// BitNetExpert158b_6743098655628
// MI455X (gfx1250) — 8-bit matrix operand order repaired and checked by reading, compiled, not run on hardware
//
#include <hip/hip_runtime.h>
#include <hip/hip_bf16.h>
#include <stdint.h>


typedef int   v16i __attribute__((ext_vector_type(16)));
typedef float v8f  __attribute__((ext_vector_type(8)));

#define THRESH 0.05f
#define FP8_ONE  0x38u
#define FP8_NEG1 0xB8u

__device__ __forceinline__ unsigned char ternary_fp8(float v) {
    return (__builtin_fabsf(v) < THRESH) ? (unsigned char)0u
         : (v > 0.f ? (unsigned char)FP8_ONE : (unsigned char)FP8_NEG1);
}

__device__ __forceinline__ void async_b128_gvs(unsigned lds_off, unsigned voff,
                                               const void* sbase) {
#if defined(__gfx1250__)
    asm volatile("global_load_async_to_lds_b128 %0, %1, %2"
                 :: "v"(lds_off), "v"(voff), "s"(sbase) : "memory");
#else
    (void)lds_off; (void)voff; (void)sbase;
#endif
}
__device__ __forceinline__ void async_b64_gvs(unsigned lds_off, unsigned voff,
                                              const void* sbase) {
#if defined(__gfx1250__)
    asm volatile("global_load_async_to_lds_b64 %0, %1, %2"
                 :: "v"(lds_off), "v"(voff), "s"(sbase) : "memory");
#else
    (void)lds_off; (void)voff; (void)sbase;
#endif
}
__device__ __forceinline__ void wait_async0() {
#if defined(__gfx1250__)
    asm volatile("s_wait_asynccnt 0" ::: "memory");
#endif
}

__global__ __launch_bounds__(256)
void quant_kernel(const float* __restrict__ X, unsigned char* __restrict__ Q, int n4) {
    int i = blockIdx.x * 256 + threadIdx.x;
    if (i >= n4) return;
    float4 v = ((const float4*)X)[i];
    uchar4 q;
    q.x = ternary_fp8(v.x); q.y = ternary_fp8(v.y);
    q.z = ternary_fp8(v.z); q.w = ternary_fp8(v.w);
    ((uchar4*)Q)[i] = q;
}

#define BM 128
#define BN 128
#define BK 128

template <int M, int N, int K>
__global__ __launch_bounds__(256)
void bitgemm_fp8(const unsigned char* __restrict__ A,
                 const unsigned char* __restrict__ Bw,
                 float* __restrict__ C) {
    __shared__ __align__(16) unsigned char sA[2][BM * BK];
    __shared__ __align__(16) unsigned char sB[2][BN * BK];

    const int tid  = threadIdx.x;
    const int lane = tid & 31;
    const int wave = tid >> 5;
    const int wm   = wave & 1;
    const int wn   = wave >> 1;
    const int m0   = blockIdx.y * BM;
    const int n0   = blockIdx.x * BN;
    const int hi   = lane >> 4;
    const int lr   = lane & 15;
    const int rx   = (lr & 7) << 4;

    const int gc = (tid & 7) * 16;
    const int gr = tid >> 3;
    const int ch = tid & 7;

    const unsigned sA0 = (unsigned)(unsigned long long)&sA[0][0];
    const unsigned sB0 = (unsigned)(unsigned long long)&sB[0][0];

    v8f acc[4][2] = {};

    auto issue_tile = [&](int buf, int k0) {
        const unsigned char* sbA = A  + k0;
        const unsigned char* sbB = Bw + k0;
        const unsigned aBuf = sA0 + (unsigned)buf * (BM * BK);
        const unsigned bBuf = sB0 + (unsigned)buf * (BN * BK);
        #pragma unroll
        for (int i = 0; i < 4; ++i) {
            const int r    = gr + 32 * i;
            const int rx16 = (r & 7) << 4;
            async_b128_gvs(bBuf + r * BK + (gc ^ rx16),
                           (unsigned)((n0 + r) * K + gc), sbB);
            const unsigned voffA = (unsigned)((m0 + r) * K + gc);
            const unsigned lo0 = ((16 * (ch >> 1)) ^ rx16) + 8 * (ch & 1);
            const unsigned lo1 = ((16 * (4 + (ch >> 1))) ^ rx16) + 8 * (ch & 1);
            async_b64_gvs(aBuf + r * BK + lo0, voffA,     sbA);
            async_b64_gvs(aBuf + r * BK + lo1, voffA + 8, sbA);
        }
    };

    constexpr int S = K / BK;
    issue_tile(0, 0);

    #pragma unroll 2
    for (int step = 0; step < S; ++step) {
        const int cur = step & 1;
        wait_async0();
        __syncthreads();
        if (step + 1 < S) issue_tile(cur ^ 1, (step + 1) * BK);

        v16i bf[2];
        #pragma unroll
        for (int j = 0; j < 2; ++j) {
            const int row = wn * 32 + j * 16 + lr;
            const unsigned char* base = &sB[cur][row * BK];
            #pragma unroll
            for (int g = 0; g < 4; ++g) {
                int4 d = *(const int4*)(base + ((32 * g + 16 * hi) ^ rx));
                bf[j][4*g+0] = d.x; bf[j][4*g+1] = d.y;
                bf[j][4*g+2] = d.z; bf[j][4*g+3] = d.w;
            }
        }

        #pragma unroll
        for (int i = 0; i < 4; ++i) {
            const int row = wm * 64 + i * 16 + lr;
            const unsigned char* base = &sA[cur][row * BK];
            v16i af;
            #pragma unroll
            for (int g = 0; g < 4; ++g) {
                int2 d0 = *(const int2*)(base + ((16 * g) ^ rx) + 8 * hi);
                int2 d1 = *(const int2*)(base + ((64 + 16 * g) ^ rx) + 8 * hi);
                af[4*g+0] = d0.x; af[4*g+1] = d0.y;
                af[4*g+2] = d1.x; af[4*g+3] = d1.y;
            }
            acc[i][0] = __builtin_amdgcn_wmma_f32_16x16x128_fp8_fp8(
                            af, bf[0], (short)0, acc[i][0], false, false);
            acc[i][1] = __builtin_amdgcn_wmma_f32_16x16x128_fp8_fp8(
                            af, bf[1], (short)0, acc[i][1], false, false);
        }
    }

    __syncthreads();
    float* sc = (wave < 4) ? (float*)(&sA[0][0]) + wave * 2048
                           : (float*)(&sB[0][0]) + (wave - 4) * 2048;
    #pragma unroll
    for (int i = 0; i < 4; ++i)
        #pragma unroll
        for (int j = 0; j < 2; ++j)
            #pragma unroll
            for (int r = 0; r < 8; ++r)
                sc[(i * 16 + 8 * hi + r) * 32 + j * 16 + lr] = acc[i][j][r];

    #pragma unroll
    for (int t = 0; t < 16; ++t) {
        float4 v = *(const float4*)(sc + t * 128 + lane * 4);
        const int row = t * 4 + (lane >> 3);
        const int col = (lane & 7) * 4;
        *(float4*)&C[(size_t)(m0 + wm * 64 + row) * N + (n0 + wn * 32 + col)] = v;
    }
}

__global__ __launch_bounds__(256)
void ln_relu_quant_kernel(const float* __restrict__ H,
                          const float* __restrict__ gamma,
                          const float* __restrict__ beta,
                          unsigned char* __restrict__ Q, int cols) {
    const int row = blockIdx.x;
    const float* h = H + (size_t)row * cols;
    float s = 0.f, ss = 0.f;
    for (int c = threadIdx.x; c < cols; c += 256) {
        float v = h[c]; s += v; ss += v * v;
    }
    __shared__ float red0[256];
    __shared__ float red1[256];
    red0[threadIdx.x] = s; red1[threadIdx.x] = ss;
    __syncthreads();
    for (int off = 128; off > 0; off >>= 1) {
        if (threadIdx.x < off) {
            red0[threadIdx.x] += red0[threadIdx.x + off];
            red1[threadIdx.x] += red1[threadIdx.x + off];
        }
        __syncthreads();
    }
    const float inv_n = 1.0f / (float)cols;
    const float mean  = red0[0] * inv_n;
    const float var   = red1[0] * inv_n - mean * mean;
    const float rs    = __frsqrt_rn(var + 1e-5f);
    unsigned char* q  = Q + (size_t)row * cols;
    for (int c = threadIdx.x; c < cols; c += 256) {
        float v = (h[c] - mean) * rs * gamma[c] + beta[c];
        v = fmaxf(v, 0.0f);
        q[c] = ternary_fp8(v);
    }
}

extern "C" void kernel_launch(void* const* d_in, const int* in_sizes, int n_in,
                              void* d_out, int out_size, void* d_ws, size_t ws_size,
                              hipStream_t stream) {
    const float* x  = (const float*)d_in[0];
    const float* W1 = (const float*)d_in[1];
    const float* g1 = (const float*)d_in[2];
    const float* b1 = (const float*)d_in[3];
    const float* W2 = (const float*)d_in[4];
    const float* g2 = (const float*)d_in[5];
    const float* b2 = (const float*)d_in[6];
    const float* W3 = (const float*)d_in[7];
    float* out = (float*)d_out;

    constexpr int B = 4096, DIN = 2048, H = 8192, DOUT = 2048;

    char* ws = (char*)d_ws;
    size_t off = 0;
    auto carve = [&](size_t bytes) {
        char* p = ws + off;
        off += (bytes + 255) & ~(size_t)255;
        return p;
    };
    unsigned char* xq  = (unsigned char*)carve((size_t)B * DIN);
    unsigned char* w1q = (unsigned char*)carve((size_t)H * DIN);
    unsigned char* w2q = (unsigned char*)carve((size_t)H * H);
    unsigned char* w3q = (unsigned char*)carve((size_t)DOUT * H);
    unsigned char* hq  = (unsigned char*)carve((size_t)B * H);
    float*         hf  = (float*)carve((size_t)B * H * sizeof(float));

    auto quant = [&](const float* src, unsigned char* dst, size_t n) {
        int n4 = (int)(n / 4);
        quant_kernel<<<(n4 + 255) / 256, 256, 0, stream>>>(src, dst, n4);
    };
    quant(x,  xq,  (size_t)B * DIN);
    quant(W1, w1q, (size_t)H * DIN);
    quant(W2, w2q, (size_t)H * H);
    quant(W3, w3q, (size_t)DOUT * H);

    bitgemm_fp8<B, H, DIN><<<dim3(H / BN, B / BM), 256, 0, stream>>>(xq, w1q, hf);
    ln_relu_quant_kernel<<<B, 256, 0, stream>>>(hf, g1, b1, hq, H);

    bitgemm_fp8<B, H, H><<<dim3(H / BN, B / BM), 256, 0, stream>>>(hq, w2q, hf);
    ln_relu_quant_kernel<<<B, 256, 0, stream>>>(hf, g2, b2, hq, H);

    bitgemm_fp8<B, DOUT, H><<<dim3(DOUT / BN, B / BM), 256, 0, stream>>>(hq, w3q, out);
}
